// QFIMetricAttention_73100343378090
// MI455X (gfx1250) — matrix operand order repaired and checked by reading, not compiled, not run on hardware
//
#include <hip/hip_runtime.h>
#include <hip/hip_bf16.h>
#include <math.h>

// ---------------------------------------------------------------------------
// Problem constants (match reference)
// ---------------------------------------------------------------------------
#define BATCH   4
#define SEQ     1024
#define DMODEL  1024
#define NHEADS  8
#define DK      128          // DMODEL / NHEADS
#define MROWS   (BATCH*SEQ)  // 4096
#define RADIUS_C 32
#define BANDW   (2*RADIUS_C+1)   // 65
#define ATT_LD  66
#define TEMP_C  0.5f
#define ALPHA_C 0.5f
#define KIND_W  0.3f
#define ENT_T   0.15f
#define GAUGE_T 0.3f

// GEMM tiling: 256 threads = 8 waves; block tile 128(M) x 32(N), K-step 64.
// Each wave owns one 16-row M-subtile and both 16-col N-subtiles:
// 4 WMMAs per barrier pair, A fragment reused across N accumulators.
#define BM 128
#define BN 32
#define BK 64
#define AS_LD 72   // BK + 8 halves pad (16B-aligned rows, de-conflicted banks)

typedef __attribute__((ext_vector_type(16))) _Float16 v16h;
typedef __attribute__((ext_vector_type(8)))  float    v8f;

// scalar block layout (floats): 0 gaugeAcc, 1 flag, 2 social, 3 penaltyScale,
// 4..7 gate[b], 8..39 Hjoint[bh], 40..71 Hq[bh], 72..103 Hk[bh]
#define SC_GAUGE   0
#define SC_FLAG    1
#define SC_SOCIAL  2
#define SC_SCALE   3
#define SC_GATE    4
#define SC_HJ      8
#define SC_HQ      40
#define SC_HK      72

// ---------------------------------------------------------------------------
// Block reductions (blockDim.x == 128, power of two)
// ---------------------------------------------------------------------------
__device__ __forceinline__ float blockReduceSum(float v, float* sm) {
  const int t = threadIdx.x;
  sm[t] = v; __syncthreads();
  for (int s = 64; s > 0; s >>= 1) { if (t < s) sm[t] += sm[t + s]; __syncthreads(); }
  float r = sm[0]; __syncthreads();
  return r;
}
__device__ __forceinline__ float blockReduceMax(float v, float* sm) {
  const int t = threadIdx.x;
  sm[t] = v; __syncthreads();
  for (int s = 64; s > 0; s >>= 1) { if (t < s) sm[t] = fmaxf(sm[t], sm[t + s]); __syncthreads(); }
  float r = sm[0]; __syncthreads();
  return r;
}

// ---------------------------------------------------------------------------
// Shared WMMA macro-tile routine.
// Computes a BM x BN tile of Y = A[M,K] @ W[N,K]^T (both K-contiguous per row
// == torch Linear weight layout / row-major activations).
// K order chosen here (A and B share it; not the ISA table order), wave32:
//   A (16x32, MxK): lane l holds row M=l&15, K-contiguous:
//       element e maps to K = 16*(l>>4) + e
//   B (32x16, KxN): lane l holds column N=l&15, K-contiguous:
//       element e maps to K = 16*(l>>4) + e
//   C/D (16x16 f32): VGPR g, lane l -> M = g + 8*(l>>4), N = l&15
// acc[0], acc[1] are the two N-subtile accumulators for this wave.
// ---------------------------------------------------------------------------
__device__ __forceinline__ void wmma_macro_tile(const _Float16* __restrict__ A,
                                                const _Float16* __restrict__ W,
                                                int Kdim, int m0, int n0,
                                                _Float16 (&Ash)[BM][AS_LD],
                                                _Float16 (&Bsh)[BN][AS_LD],
                                                v8f& acc0, v8f& acc1)
{
  const int t    = threadIdx.x;
  const int wave = t >> 5;
  const int lane = t & 31;
  const int khiA = (lane >> 4) * 16;
  const int khiB = (lane >> 4) * 16;

  for (int k0 = 0; k0 < Kdim; k0 += BK) {
    // stage A tile: BM x BK halves = 1024 uint4, 4 per thread, coalesced
#pragma unroll
    for (int it = 0; it < 4; ++it) {
      const int idx = t + it * 256;
      const int r = idx >> 3, c4 = idx & 7;
      *(uint4*)(&Ash[r][c4 * 8]) =
          *(const uint4*)(A + (size_t)(m0 + r) * Kdim + k0 + c4 * 8);
    }
    // stage B tile: BN x BK halves = 256 uint4, 1 per thread
    {
      const int r = t >> 3, c4 = t & 7;
      *(uint4*)(&Bsh[r][c4 * 8]) =
          *(const uint4*)(W + (size_t)(n0 + r) * Kdim + k0 + c4 * 8);
    }
    __syncthreads();
#pragma unroll
    for (int kk = 0; kk < BK; kk += 32) {
      const _Float16* As = &Ash[wave * 16 + (lane & 15)][kk];
      v16h a;
#pragma unroll
      for (int e = 0; e < 16; ++e)
        a[e] = As[khiA + e];

      const _Float16* Bs0 = &Bsh[(lane & 15)][kk];
      const _Float16* Bs1 = &Bsh[16 + (lane & 15)][kk];
      v16h b0, b1;
#pragma unroll
      for (int e = 0; e < 16; ++e) { b0[e] = Bs0[khiB + e]; b1[e] = Bs1[khiB + e]; }

      acc0 = __builtin_amdgcn_wmma_f32_16x16x32_f16(false, a, false, b0,
                                                    (short)0, acc0, false, false);
      acc1 = __builtin_amdgcn_wmma_f32_16x16x32_f16(false, a, false, b1,
                                                    (short)0, acc1, false, false);
    }
    __syncthreads();
  }
}

// Generic linear layer: Y = scale * (A @ W^T) + bias, optional ReLU.
__global__ __launch_bounds__(256)
void gemm_wmma_f16(const _Float16* __restrict__ A, const _Float16* __restrict__ W,
                   const float* __restrict__ bias, const float* __restrict__ scale,
                   float* __restrict__ Y, int N, int Kdim, int relu)
{
  __shared__ __align__(16) _Float16 Ash[BM][AS_LD];
  __shared__ __align__(16) _Float16 Bsh[BN][AS_LD];
  const int n0 = blockIdx.x * BN;
  const int m0 = blockIdx.y * BM;
  const int lane = threadIdx.x & 31;
  const int wave = threadIdx.x >> 5;

  v8f acc0 = {}, acc1 = {};
  wmma_macro_tile(A, W, Kdim, m0, n0, Ash, Bsh, acc0, acc1);

  const float s = scale ? *scale : 1.0f;
  const int mBase = m0 + wave * 16 + ((lane >> 4) << 3);
#pragma unroll
  for (int ns = 0; ns < 2; ++ns) {
    const v8f& c = ns ? acc1 : acc0;
    const int nn = n0 + ns * 16 + (lane & 15);
    const float bv = bias ? bias[nn] : 0.0f;
#pragma unroll
    for (int g = 0; g < 8; ++g) {
      float v = s * c[g] + bv;
      if (relu) v = fmaxf(v, 0.0f);
      Y[(size_t)(mBase + g) * N + nn] = v;
    }
  }
}

// joint = Qs @ Ks^T per (b,h); accumulate H_joint = sum(-p*log(p+1e-10))
// without materializing the SxS joint matrix.
__global__ __launch_bounds__(256)
void joint_entropy_wmma(const _Float16* __restrict__ Qs,
                        const _Float16* __restrict__ Ks,
                        float* __restrict__ scal)
{
  __shared__ __align__(16) _Float16 Ash[BM][AS_LD];
  __shared__ __align__(16) _Float16 Bsh[BN][AS_LD];
  __shared__ float red[8];
  const int bh = blockIdx.z;
  const _Float16* A = Qs + (size_t)bh * SEQ * DK;
  const _Float16* W = Ks + (size_t)bh * SEQ * DK;
  const int n0 = blockIdx.x * BN;
  const int m0 = blockIdx.y * BM;
  const int lane = threadIdx.x & 31;
  const int wave = threadIdx.x >> 5;

  v8f acc0 = {}, acc1 = {};
  wmma_macro_tile(A, W, DK, m0, n0, Ash, Bsh, acc0, acc1);

  float ent = 0.0f;
#pragma unroll
  for (int g = 0; g < 8; ++g) {
    const float p0 = acc0[g], p1 = acc1[g];
    ent += -p0 * logf(p0 + 1e-10f);
    ent += -p1 * logf(p1 + 1e-10f);
  }
  for (int off = 16; off > 0; off >>= 1) ent += __shfl_down(ent, off, 32);
  if (lane == 0) red[wave] = ent;
  __syncthreads();
  if (threadIdx.x == 0) {
    float s = 0.f;
#pragma unroll
    for (int w = 0; w < 8; ++w) s += red[w];
    atomicAdd(&scal[SC_HJ + bh], s);
  }
}

// ---------------------------------------------------------------------------
// Small helper kernels
// ---------------------------------------------------------------------------
__global__ void init_scalars_kernel(float* scal) {
  if (threadIdx.x < 128) scal[threadIdx.x] = 0.0f;
}

__global__ void cvt_f16_kernel(const float* __restrict__ s,
                               _Float16* __restrict__ d, size_t n) {
  for (size_t i = (size_t)blockIdx.x * blockDim.x + threadIdx.x; i < n;
       i += (size_t)gridDim.x * blockDim.x)
    d[i] = (_Float16)s[i];
}

__global__ __launch_bounds__(128)
void gauge_kernel(const float* __restrict__ x, const float* __restrict__ inv,
                  float* scal) {
  __shared__ float red[128];
  const int row = blockIdx.x, t = threadIdx.x;
  float r2 = 0.f, x2 = 0.f;
  for (int c = t; c < DMODEL; c += 128) {
    const size_t o = (size_t)row * DMODEL + c;
    const float xv = x[o], dv = xv - inv[o];
    r2 += dv * dv; x2 += xv * xv;
  }
  r2 = blockReduceSum(r2, red);
  x2 = blockReduceSum(x2, red);
  if (t == 0)
    atomicAdd(&scal[SC_GAUGE],
              (sqrtf(r2) / (sqrtf(x2) + 1e-8f)) * (1.0f / (float)MROWS));
}

__global__ void flag_kernel(float* scal) {
  if (threadIdx.x == 0) scal[SC_FLAG] = (scal[SC_GAUGE] > GAUGE_T) ? 1.0f : 0.0f;
}

__global__ void select_cvt_kernel(const float* __restrict__ x,
                                  const float* __restrict__ inv,
                                  const float* __restrict__ scal,
                                  _Float16* __restrict__ dst, size_t n) {
  const float f = scal[SC_FLAG];
  for (size_t i = (size_t)blockIdx.x * blockDim.x + threadIdx.x; i < n;
       i += (size_t)gridDim.x * blockDim.x)
    dst[i] = (_Float16)(f > 0.5f ? inv[i] : x[i]);
}

// row softmax over head dim (dk=128): [B,S,D] f32 -> [B*H,S,DK] f16
__global__ __launch_bounds__(128)
void head_rowsoftmax(const float* __restrict__ X, _Float16* __restrict__ P) {
  __shared__ float red[128];
  const int i = blockIdx.x, bh = blockIdx.y;
  const int b = bh / NHEADS, h = bh % NHEADS, t = threadIdx.x;
  const float v = X[((size_t)(b * SEQ + i)) * DMODEL + h * DK + t];
  const float m = blockReduceMax(v, red);
  const float e = expf(v - m);
  const float z = blockReduceSum(e, red);
  P[((size_t)bh * SEQ + i) * DK + t] = (_Float16)(e / z);
}

// H_q, H_k: entropy of softmax(column-sums of Q/K over S)
__global__ __launch_bounds__(128)
void head_marginal_entropy(const float* __restrict__ Q,
                           const float* __restrict__ Kf, float* scal) {
  __shared__ float red[128];
  const int bh = blockIdx.x, b = bh / NHEADS, h = bh % NHEADS, t = threadIdx.x;
  float sq = 0.f, sk = 0.f;
  for (int i = 0; i < SEQ; ++i) {
    const size_t o = ((size_t)(b * SEQ + i)) * DMODEL + h * DK + t;
    sq += Q[o]; sk += Kf[o];
  }
  float m = blockReduceMax(sq, red);
  float e = expf(sq - m);
  float z = blockReduceSum(e, red);
  float p = e / z;
  const float hq = blockReduceSum(-p * logf(p + 1e-10f), red);
  m = blockReduceMax(sk, red);
  e = expf(sk - m);
  z = blockReduceSum(e, red);
  p = e / z;
  const float hk = blockReduceSum(-p * logf(p + 1e-10f), red);
  if (t == 0) { scal[SC_HQ + bh] = hq; scal[SC_HK + bh] = hk; }
}

__global__ void gate_kernel(float* scal) {
  const int b = threadIdx.x;
  if (b < BATCH) {
    float s = 0.f;
    for (int h = 0; h < NHEADS; ++h) {
      const int bh = b * NHEADS + h;
      s += scal[SC_HJ + bh] - scal[SC_HQ + bh] - scal[SC_HK + bh];
    }
    s /= (float)NHEADS;
    scal[SC_GATE + b] = (s > ENT_T) ? 1.0f : 0.0f;
  }
}

// banded Bures-distance attention weights (softmax over 65-wide band)
__global__ __launch_bounds__(128)
void attn_scores_kernel(const float* __restrict__ Q, const float* __restrict__ Kf,
                        float* __restrict__ attn) {
  __shared__ float qsh[DK];
  __shared__ float red[128];
  const int i = blockIdx.x, bh = blockIdx.y;
  const int b = bh / NHEADS, h = bh % NHEADS, t = threadIdx.x;
  const float qv = Q[((size_t)(b * SEQ + i)) * DMODEL + h * DK + t];
  qsh[t] = qv;
  const float q2 = blockReduceSum(qv * qv, red);   // syncs -> qsh visible
  const int jj = t;
  const int j = i - RADIUS_C + jj;
  const bool valid = (jj < BANDW) && (j >= 0) && (j < SEQ);
  float score = -1e30f;
  if (valid) {
    const size_t rk = ((size_t)(b * SEQ + j)) * DMODEL + h * DK;
    float dot = 0.f, k2 = 0.f;
    for (int d = 0; d < DK; ++d) {
      const float kv = Kf[rk + d];
      dot += qsh[d] * kv; k2 += kv * kv;
    }
    const float qn = fmaxf(sqrtf(q2), 1e-12f);
    const float kn = fmaxf(sqrtf(k2), 1e-12f);
    const float sim = dot / (qn * kn);
    const float dist = sqrtf(fminf(fmaxf(2.0f * (1.0f - sim), 0.0f), 4.0f));
    score = (-ALPHA_C * dist) / TEMP_C;
  }
  const float m = blockReduceMax(score, red);
  const float w = valid ? expf(score - m) : 0.0f;
  const float z = blockReduceSum(w, red);
  if (jj < BANDW)
    attn[((size_t)bh * SEQ + i) * ATT_LD + jj] = w / z;
}

// out = gate[b] * (attn @ V); write f16 for downstream WMMA GEMMs
__global__ __launch_bounds__(128)
void attn_v_kernel(const float* __restrict__ attn, const float* __restrict__ V,
                   const float* __restrict__ scal, _Float16* __restrict__ outh) {
  const int i = blockIdx.x, bh = blockIdx.y;
  const int b = bh / NHEADS, h = bh % NHEADS, t = threadIdx.x;  // t = d
  const float gate = scal[SC_GATE + b];
  float acc = 0.f;
  const size_t ab = ((size_t)bh * SEQ + i) * ATT_LD;
  for (int jj = 0; jj < BANDW; ++jj) {
    const int j = i - RADIUS_C + jj;
    if (j < 0 || j >= SEQ) continue;
    acc += attn[ab + jj] * V[((size_t)(b * SEQ + j)) * DMODEL + h * DK + t];
  }
  outh[((size_t)(b * SEQ + i)) * DMODEL + h * DK + t] = (_Float16)(acc * gate);
}

// curv = sigmoid(h1 @ W2^T + b2); accumulate mean into social
__global__ __launch_bounds__(128)
void curvature_kernel(const float* __restrict__ h1, const float* __restrict__ W2,
                      const float* __restrict__ b2, float* scal) {
  __shared__ float red[128];
  const int row = blockIdx.x, t = threadIdx.x;
  float d = 0.f;
  for (int c = t; c < DMODEL / 2; c += 128)
    d += h1[(size_t)row * (DMODEL / 2) + c] * W2[c];
  const float v = blockReduceSum(d, red);
  if (t == 0) {
    const float sg = 1.0f / (1.0f + expf(-(v + b2[0])));
    atomicAdd(&scal[SC_SOCIAL], sg * (1.0f / (float)MROWS));
  }
}

__global__ void penalty_kernel(float* scal) {
  if (threadIdx.x == 0) {
    float pen = expf(-scal[SC_SOCIAL] / TEMP_C);
    pen = fminf(fmaxf(pen, 0.5f), 1.5f);
    scal[SC_SCALE] = 1.0f - KIND_W * (1.0f - pen);
  }
}

// ---------------------------------------------------------------------------
// Host-side orchestration
// ---------------------------------------------------------------------------
extern "C" void kernel_launch(void* const* d_in, const int* in_sizes, int n_in,
                              void* d_out, int out_size, void* d_ws, size_t ws_size,
                              hipStream_t stream) {
  (void)in_sizes; (void)n_in; (void)out_size; (void)ws_size;
  const float* x  = (const float*)d_in[0];
  const float* Wq = (const float*)d_in[1];  const float* bq = (const float*)d_in[2];
  const float* Wk = (const float*)d_in[3];  const float* bk = (const float*)d_in[4];
  const float* Wv = (const float*)d_in[5];  const float* bv = (const float*)d_in[6];
  const float* Wo = (const float*)d_in[7];  const float* bo = (const float*)d_in[8];
  const float* Wp = (const float*)d_in[9];  const float* bp = (const float*)d_in[10];
  const float* W1 = (const float*)d_in[11]; const float* b1 = (const float*)d_in[12];
  const float* W2 = (const float*)d_in[13]; const float* b2 = (const float*)d_in[14];

  // bump allocator over workspace (256B aligned)
  char* p = (char*)d_ws;
  auto alloc = [&](size_t bytes) -> char* {
    char* r = p;
    p += (bytes + 255) & ~(size_t)255;
    return r;
  };
  const size_t nX = (size_t)MROWS * DMODEL;
  _Float16* xh   = (_Float16*)alloc(nX * 2);
  _Float16* Wqh  = (_Float16*)alloc((size_t)DMODEL * DMODEL * 2);
  _Float16* Wkh  = (_Float16*)alloc((size_t)DMODEL * DMODEL * 2);
  _Float16* Wvh  = (_Float16*)alloc((size_t)DMODEL * DMODEL * 2);
  _Float16* Woh  = (_Float16*)alloc((size_t)DMODEL * DMODEL * 2);
  _Float16* Wph  = (_Float16*)alloc((size_t)DMODEL * DMODEL * 2);
  _Float16* W1h  = (_Float16*)alloc((size_t)(DMODEL / 2) * DMODEL * 2);
  float*    inv  = (float*)alloc(nX * 4);
  _Float16* xqkh = (_Float16*)alloc(nX * 2);          // later reused as outh
  float*    Qf   = (float*)alloc(nX * 4);
  float*    Kf   = (float*)alloc(nX * 4);
  float*    Vf   = (float*)alloc(nX * 4);             // later reused as h1
  _Float16* Qs   = (_Float16*)alloc(nX * 2);
  _Float16* Ks   = (_Float16*)alloc(nX * 2);
  float*    attn = (float*)alloc((size_t)BATCH * NHEADS * SEQ * ATT_LD * 4);
  float*    scal = (float*)alloc(512);
  _Float16* outh = xqkh;       // x_qk no longer needed once Q,K are formed
  float*    h1   = Vf;         // V no longer needed after attn@V

  const dim3 blk128(128);
  const dim3 blk256(256);
  const dim3 gemmFull(DMODEL / BN, MROWS / BM);       // N=1024 GEMMs

  init_scalars_kernel<<<1, 128, 0, stream>>>(scal);

  // fp32 -> fp16 staging
  cvt_f16_kernel<<<2048, 256, 0, stream>>>(x,  xh,  nX);
  cvt_f16_kernel<<<1024, 256, 0, stream>>>(Wq, Wqh, (size_t)DMODEL * DMODEL);
  cvt_f16_kernel<<<1024, 256, 0, stream>>>(Wk, Wkh, (size_t)DMODEL * DMODEL);
  cvt_f16_kernel<<<1024, 256, 0, stream>>>(Wv, Wvh, (size_t)DMODEL * DMODEL);
  cvt_f16_kernel<<<1024, 256, 0, stream>>>(Wo, Woh, (size_t)DMODEL * DMODEL);
  cvt_f16_kernel<<<1024, 256, 0, stream>>>(Wp, Wph, (size_t)DMODEL * DMODEL);
  cvt_f16_kernel<<<512,  256, 0, stream>>>(W1, W1h, (size_t)(DMODEL / 2) * DMODEL);

  // gauge-invariance test
  gemm_wmma_f16<<<gemmFull, blk256, 0, stream>>>(xh, Wph, bp, nullptr, inv,
                                                 DMODEL, DMODEL, 0);
  gauge_kernel<<<MROWS, blk128, 0, stream>>>(x, inv, scal);
  flag_kernel<<<1, 32, 0, stream>>>(scal);
  select_cvt_kernel<<<2048, 256, 0, stream>>>(x, inv, scal, xqkh, nX);

  // Q, K (from x_qk) and V (from x)
  gemm_wmma_f16<<<gemmFull, blk256, 0, stream>>>(xqkh, Wqh, bq, nullptr, Qf,
                                                 DMODEL, DMODEL, 0);
  gemm_wmma_f16<<<gemmFull, blk256, 0, stream>>>(xqkh, Wkh, bk, nullptr, Kf,
                                                 DMODEL, DMODEL, 0);
  gemm_wmma_f16<<<gemmFull, blk256, 0, stream>>>(xh, Wvh, bv, nullptr, Vf,
                                                 DMODEL, DMODEL, 0);

  // entanglement entropy gate
  head_rowsoftmax<<<dim3(SEQ, BATCH * NHEADS), blk128, 0, stream>>>(Qf, Qs);
  head_rowsoftmax<<<dim3(SEQ, BATCH * NHEADS), blk128, 0, stream>>>(Kf, Ks);
  joint_entropy_wmma<<<dim3(SEQ / BN, SEQ / BM, BATCH * NHEADS), blk256, 0,
                       stream>>>(Qs, Ks, scal);
  head_marginal_entropy<<<BATCH * NHEADS, blk128, 0, stream>>>(Qf, Kf, scal);
  gate_kernel<<<1, 32, 0, stream>>>(scal);

  // banded QFI attention
  attn_scores_kernel<<<dim3(SEQ, BATCH * NHEADS), blk128, 0, stream>>>(Qf, Kf,
                                                                       attn);
  attn_v_kernel<<<dim3(SEQ, BATCH * NHEADS), blk128, 0, stream>>>(attn, Vf,
                                                                  scal, outh);

  // kindness penalty (impact head)
  gemm_wmma_f16<<<dim3((DMODEL / 2) / BN, MROWS / BM), blk256, 0, stream>>>(
      outh, W1h, b1, nullptr, h1, DMODEL / 2, DMODEL, 1 /*relu*/);
  curvature_kernel<<<MROWS, blk128, 0, stream>>>(h1, W2, b2, scal);
  penalty_kernel<<<1, 32, 0, stream>>>(scal);

  // final projection with penalty scale folded into the GEMM epilogue
  gemm_wmma_f16<<<gemmFull, blk256, 0, stream>>>(outh, Woh, bo, &scal[SC_SCALE],
                                                 (float*)d_out, DMODEL, DMODEL,
                                                 0);
}
